// MaskGIT_Template_10986526343952
// MI455X (gfx1250) — compile-verified, matrix operand order checked by reading, not run on hardware
//
#include <hip/hip_runtime.h>
#include <hip/hip_bf16.h>
#include <math.h>

typedef __attribute__((ext_vector_type(16))) _Float16 v16h;
typedef __attribute__((ext_vector_type(8)))  float    v8f;

#define TT 11
#define BB 32
#define NN 256
#define VV 1024
#define CC 256

#if defined(__has_builtin)
#if __has_builtin(__builtin_amdgcn_global_load_async_to_lds_b128) && \
    __has_builtin(__builtin_amdgcn_global_store_async_from_lds_b128) && \
    __has_builtin(__builtin_amdgcn_s_wait_asynccnt)
#define USE_ASYNC_LDS 1
#endif
#endif

#ifdef USE_ASYNC_LDS
// The builtins take pointers to GCC-style v4i vectors:
//   arg0: v4i addrspace(1)*  (global)   arg1: v4i addrspace(3)* (LDS)
//   arg2: imm byte offset (applies to BOTH addresses)   arg3: imm cpol
typedef int v4i_gcc __attribute__((vector_size(16)));
typedef __attribute__((address_space(1))) v4i_gcc gas_v4i;   // global
typedef __attribute__((address_space(3))) v4i_gcc las_v4i;   // LDS
#endif

__device__ __forceinline__ unsigned hash_u32(unsigned x) {
    x ^= x >> 17; x *= 0xed5ad4bbu;
    x ^= x >> 11; x *= 0xac4c1b51u;
    x ^= x >> 15; x *= 0x31848babu;
    x ^= x >> 14;
    return x;
}

__device__ __forceinline__ float gumbel_from_hash(unsigned h) {
    // u in (0,1), then standard Gumbel(0,1)
    float u = ((float)((h >> 8) + 1u)) * (1.0f / 16777218.0f);
    return -__logf(-__logf(u));
}

__global__ void maskgit_init(int* __restrict__ cur) {
    int i = blockIdx.x * blockDim.x + threadIdx.x;
    if (i < BB * NN) cur[i] = VV;   // all tokens start masked
}

__global__ __launch_bounds__(512) void maskgit_step(
    const float* __restrict__ logits,  // [T,B,N,V]
    int*   __restrict__ cur_ids,       // [B,N]   (ws)
    int*   __restrict__ seq,           // [B,T,N] (ws)
    float* __restrict__ masks,         // [B,T,N] (out)
    int t)
{
    __shared__ float s_m[NN], s_la[NN], s_sum[NN], s_conf[NN], s_sort[NN];
    __shared__ int   s_idx[NN], s_samp[NN];
    __shared__ int   s_cnt;

    const int b    = blockIdx.x;
    const int tid  = threadIdx.x;
    const int lane = tid & 31;
    const int wave = tid >> 5;          // 0..15, one 16-token tile per wave

    if (tid == 0) s_cnt = 0;

    const float ratio = (float)(t + 1) / (float)TT;
    const float temp  = 4.5f * (1.0f - ratio);   // CHOICE_TEMPERATURE * (1 - ratio)

    const float* lbase = logits + (((size_t)t * BB + b) * NN) * VV;

    // ---------- Pass 1: per-token max + Gumbel-argmax (categorical sample) ----------
    for (int tok = 0; tok < 16; ++tok) {
        const int n = wave * 16 + tok;
        const float* lrow = lbase + (size_t)n * VV;
        float lm = -INFINITY, bs = -INFINITY, bl = 0.0f;
        int   bi = 0;
        const unsigned base_h = ((unsigned)t * 0x9E3779B9u) ^
                                ((unsigned)(b * NN + n) * 0x85EBCA6Bu);
        for (int v = lane; v < VV; v += 32) {
            float x = lrow[v];
            __builtin_prefetch(lrow + v + 256, 0, 1);   // global_prefetch_b8
            lm = fmaxf(lm, x);
            float g = gumbel_from_hash(hash_u32(base_h + (unsigned)v));
            float s = x + g;
            if (s > bs) { bs = s; bl = x; bi = v; }
        }
        #pragma unroll
        for (int off = 16; off >= 1; off >>= 1) {
            lm = fmaxf(lm, __shfl_xor(lm, off, 32));
            float os = __shfl_xor(bs, off, 32);
            float ol = __shfl_xor(bl, off, 32);
            int   oi = __shfl_xor(bi, off, 32);
            if (os > bs) { bs = os; bl = ol; bi = oi; }
        }
        if (lane == 0) { s_m[n] = lm; s_la[n] = bl; s_idx[n] = bi; }
    }
    __syncthreads();

    // ---------- Pass 2: softmax partition function via WMMA (ones-matrix trick) ----
    // A = 16x32 tile of exp(x - max) per wave, B = ones => every C column = row sum.
    v16h bones;
    #pragma unroll
    for (int i = 0; i < 16; ++i) bones[i] = (_Float16)1.0f;

    v8f acc = {};
    const int row  = lane & 15;
    const int nrow = wave * 16 + row;
    const float  m    = s_m[nrow];
    const float* lrow = lbase + (size_t)nrow * VV;
    const int khi = (lane >= 16) ? 8 : 0;   // A-matrix K split per ISA layout

    for (int k0 = 0; k0 < VV; k0 += 64) {   // 2 WMMAs per iteration
        v16h a0, a1;
        #pragma unroll
        for (int i = 0; i < 16; ++i) {
            // lanes 0-15: K = {0..7, 16..23}; lanes 16-31: K = {8..15, 24..31}
            int koff = ((i < 8) ? i : (i + 8)) + khi;
            a0[i] = (_Float16)__expf(lrow[k0      + koff] - m);
            a1[i] = (_Float16)__expf(lrow[k0 + 32 + koff] - m);
        }
        acc = __builtin_amdgcn_wmma_f32_16x16x32_f16(
                  false, a0, false, bones, (short)0, acc, false, false);
        acc = __builtin_amdgcn_wmma_f32_16x16x32_f16(
                  false, a1, false, bones, (short)0, acc, false, false);
    }
    // C/D layout: lanes 0-15 hold rows 0..7 in vgpr 0..7; lanes 16-31 hold rows 8..15
    if (lane == 0) {
        #pragma unroll
        for (int j = 0; j < 8; ++j) s_sum[wave * 16 + j] = acc[j];
    } else if (lane == 16) {
        #pragma unroll
        for (int j = 0; j < 8; ++j) s_sum[wave * 16 + 8 + j] = acc[j];
    }
    __syncthreads();

    // ---------- Phase 3: confidences, mask record, sampled ids ----------
    if (tid < NN) {
        const int n = tid;
        int  cur     = cur_ids[b * NN + n];
        bool unknown = (cur == VV);
        int  samp    = unknown ? s_idx[n] : cur;
        float conf;
        if (unknown) {
            // log p(sampled) = logit[argmax] - max - log(sum exp)
            float logsel = s_la[n] - s_m[n] - __logf(s_sum[n]);
            unsigned h2 = hash_u32(((unsigned)t * 0xC2B2AE35u) ^
                                   (unsigned)(b * NN + n) ^ 0xDEADBEEFu);
            conf = logsel + temp * gumbel_from_hash(h2);
            atomicAdd(&s_cnt, 1);
        } else {
            conf = INFINITY;   // known tokens never re-masked
        }
        s_conf[n] = conf;
        s_sort[n] = conf;
        s_samp[n] = samp;
        masks[((size_t)b * TT + t) * NN + n] = unknown ? 1.0f : 0.0f;
        seq  [((size_t)b * TT + t) * NN + n] = samp;
    }
    __syncthreads();

    // ---------- Phase 4: bitonic sort (ascending) of 256 confidences ----------
    for (int k = 2; k <= NN; k <<= 1) {
        for (int j = k >> 1; j > 0; j >>= 1) {
            if (tid < NN) {
                int ixj = tid ^ j;
                if (ixj > tid) {
                    float a0 = s_sort[tid], b0 = s_sort[ixj];
                    bool up = ((tid & k) == 0);
                    if ((a0 > b0) == up) { s_sort[tid] = b0; s_sort[ixj] = a0; }
                }
            }
            __syncthreads();
        }
    }

    // ---------- Phase 5: cutoff + state update ----------
    int num_unknown = s_cnt;
    float mlf = floorf((float)NN * __cosf(ratio * 1.57079632679f));
    mlf = fmaxf(1.0f, fminf((float)num_unknown - 1.0f, mlf));
    int ml = (int)mlf;
    float cutoff = s_sort[ml];
    if (tid < NN) {
        const int n = tid;
        bool masking = s_conf[n] < cutoff;
        cur_ids[b * NN + n] = masking ? VV : s_samp[n];
    }
}

__global__ __launch_bounds__(256) void maskgit_gather(
    const float* __restrict__ codebook,  // [V+1, C]
    const int*   __restrict__ seq,       // [B,T,N] flat == quant token order
    float*       __restrict__ quant)     // [B,T,N,C]
{
    const int lane = threadIdx.x & 31;
    const int wave = threadIdx.x >> 5;
    const size_t tok = (size_t)blockIdx.x * 8 + wave;   // grid sized exactly
    int id = seq[tok];

#ifdef USE_ASYNC_LDS
    // Stage codebook row through LDS with the gfx1250 async-LDS engines.
    // The imm offset is added to the LDS and global address alike, i.e. it
    // advances BOTH addresses; two B128 ops per lane cover 256 floats/row.
    __shared__ float buf[8 * CC];                       // 8 KB
    const float* srcp = codebook + (size_t)id * CC + lane * 4;
    float*       ldsp = &buf[wave * CC + lane * 4];
    float*       dstp = quant + tok * CC + lane * 4;

    __builtin_amdgcn_global_load_async_to_lds_b128(
        (gas_v4i*)srcp, (las_v4i*)ldsp, 0, 0);
    __builtin_amdgcn_global_load_async_to_lds_b128(
        (gas_v4i*)srcp, (las_v4i*)ldsp, 512, 0);
    __builtin_amdgcn_s_wait_asynccnt(0);                // loads landed in LDS

    __builtin_amdgcn_global_store_async_from_lds_b128(
        (gas_v4i*)dstp, (las_v4i*)ldsp, 0, 0);
    __builtin_amdgcn_global_store_async_from_lds_b128(
        (gas_v4i*)dstp, (las_v4i*)ldsp, 512, 0);
    __builtin_amdgcn_s_wait_asynccnt(0);                // drain before reuse/exit
#else
    const float4* src = (const float4*)(codebook + (size_t)id * CC);
    float4*       dst = (float4*)(quant + tok * CC);
    dst[lane]      = src[lane];        // 2 x float4 per lane covers C=256 floats
    dst[lane + 32] = src[lane + 32];
#endif
}

extern "C" void kernel_launch(void* const* d_in, const int* in_sizes, int n_in,
                              void* d_out, int out_size, void* d_ws, size_t ws_size,
                              hipStream_t stream) {
    (void)in_sizes; (void)n_in; (void)out_size; (void)ws_size;
    const float* logits   = (const float*)d_in[0];   // [T,B,N,V] f32
    const float* codebook = (const float*)d_in[1];   // [V+1,C]  f32

    float* quant = (float*)d_out;                              // [B,T,N,C]
    float* masks = quant + (size_t)BB * TT * NN * CC;          // [B,T,N]

    int* cur = (int*)d_ws;            // [B,N]
    int* seq = cur + BB * NN;         // [B,T,N]

    maskgit_init<<<(BB * NN + 255) / 256, 256, 0, stream>>>(cur);
    for (int t = 0; t < TT; ++t) {
        maskgit_step<<<BB, 512, 0, stream>>>(logits, cur, seq, masks, t);
    }
    maskgit_gather<<<(BB * TT * NN) / 8, 256, 0, stream>>>(codebook, seq, quant);
}
